// DecomposedPromptPool_12652973654374
// MI455X (gfx1250) — matrix operand order repaired and checked by reading, not compiled, not run on hardware
//
#include <hip/hip_runtime.h>
#include <hip/hip_bf16.h>
#include <math.h>
#include <stdint.h>

// ---------------------------------------------------------------------------
// DecomposedPromptPool for MI455X (gfx1250, wave32, WMMA)
//   B=16384, N=4096, D=1024, K=8 (from reference setup)
// ---------------------------------------------------------------------------

typedef __attribute__((ext_vector_type(16))) _Float16 v16h;
typedef __attribute__((ext_vector_type(8)))  _Float16 half8;
typedef __attribute__((ext_vector_type(8)))  float    v8f;
typedef __attribute__((ext_vector_type(4)))  float    v4f;

#define D_DIM 1024
#define TOPK  8
#define EPSN  1e-8f

// ---------------------------------------------------------------------------
// Kernel 0a: partial column sums of component_attention [N, D]
// ---------------------------------------------------------------------------
__global__ __launch_bounds__(256) void attn_partial_kernel(
    const float* __restrict__ attn, float* __restrict__ partial, int N) {
  const int d = blockIdx.x * 256 + threadIdx.x;
  const int c = blockIdx.y;
  const int rows = N / 32;
  const int r0 = c * rows;
  float s = 0.f;
  for (int r = 0; r < rows; ++r)
    s += attn[(size_t)(r0 + r) * D_DIM + d];
  partial[(size_t)c * D_DIM + d] = s;
}

__global__ __launch_bounds__(256) void attn_reduce_kernel(
    const float* __restrict__ partial, float* __restrict__ mean, int N) {
  const int d = blockIdx.x * 256 + threadIdx.x;
  float s = 0.f;
#pragma unroll
  for (int c = 0; c < 32; ++c) s += partial[(size_t)c * D_DIM + d];
  mean[d] = s / (float)N;
}

// ---------------------------------------------------------------------------
// Kernel 1: row-normalize (optionally pre-scaled by mean[]) -> f16 rows.
// ---------------------------------------------------------------------------
__global__ __launch_bounds__(256) void rownorm_f16_kernel(
    const float* __restrict__ x, const float* __restrict__ scale,
    _Float16* __restrict__ out) {
  const int row = blockIdx.x;
  const int tid = threadIdx.x;
  const float* xr = x + (size_t)row * D_DIM;

  float v[4];
  float ss = 0.f;
#pragma unroll
  for (int i = 0; i < 4; ++i) {
    const int d = tid + i * 256;
    float t = xr[d];
    if (scale) t *= scale[d];
    v[i] = t;
    ss += t * t;
  }

  __shared__ float red[256];
  red[tid] = ss;
  __syncthreads();
  for (int s = 128; s > 0; s >>= 1) {
    if (tid < s) red[tid] += red[tid + s];
    __syncthreads();
  }
  const float inv = 1.f / fmaxf(sqrtf(red[0]), EPSN);

  _Float16* orow = out + (size_t)row * D_DIM;
#pragma unroll
  for (int i = 0; i < 4; ++i) {
    const int d = tid + i * 256;
    orow[d] = (_Float16)(v[i] * inv);
  }
}

// ---------------------------------------------------------------------------
// Kernel 2: fused cosine-sim GEMM (f16 WMMA, f32 accum) + running top-8 +
// softmax.
//   Block = 256 threads = 8 waves; each wave owns 16 M-rows -> 128 rows/block.
//   A strip (128x1024 f16 = 256 KB) staged ONCE via async LDS DMA
//   (global_load_async_to_lds_b128, ASYNCcnt).  Each wave processes TWO
//   16-col N-tiles per iteration (2 accumulators) so every A fragment feeds
//   2 WMMAs.
// ---------------------------------------------------------------------------
__global__ __launch_bounds__(256) void sims_topk_kernel(
    const _Float16* __restrict__ aq,    // [B, 1024] normalized, f16
    const _Float16* __restrict__ kn,    // [N, 1024] normalized, f16
    float* __restrict__ topw,           // [B, 8] softmax weights (desc order)
    int* __restrict__ topi,             // [B, 8] indices (desc value order)
    float* __restrict__ outIdx,         // d_out index region, as float
    int Ncols) {
  extern __shared__ char smem[];
  _Float16* ldsA  = (_Float16*)smem;                      // 128*1024 f16 = 256 KB
  float*    stage = (float*)(smem + 128 * D_DIM * 2);     // 8 waves * 512 f32 = 16 KB

  const int tid  = threadIdx.x;
  const int w    = tid >> 5;
  const int lane = tid & 31;
  const int mBlock = blockIdx.x * 128;

  // ---- async DMA of the 128x1024 f16 A strip into LDS (ASYNCcnt path) ----
  {
    const unsigned ldsBase = (unsigned)(uintptr_t)ldsA;
    const uint64_t gBase = (uint64_t)(const void*)(aq + (size_t)mBlock * D_DIM);
    const int total = 128 * D_DIM * 2 / 16;  // 16384 x 16-byte chunks
    for (int i = tid; i < total; i += 256) {
      const unsigned ldsOff = ldsBase + (unsigned)i * 16u;
      const uint64_t gaddr = gBase + (uint64_t)i * 16u;
      asm volatile("global_load_async_to_lds_b128 %0, %1, off"
                   :: "v"(ldsOff), "v"(gaddr)
                   : "memory");
    }
    asm volatile("s_wait_asynccnt 0x0" ::: "memory");
  }
  __syncthreads();

  // Fragment layouts used by this code (A and B share one contiguous k order):
  //  A 16x32 f16: lanes 0-15 -> M=lane,   v0..7=K[kc..kc+15]
  //               lanes16-31 -> M=lane-16,v0..7=K[kc+16..kc+31]
  //  B 32x16 f16: lanes 0-15 -> N=lane,    v0..7=K[kc..kc+15]
  //               lanes16-31 -> N=lane-16, v0..7=K[kc+16..kc+31]
  //  C/D 16x16:   v[r]: lanes0-15 M=r,N=lane ; lanes16-31 M=r+8,N=lane-16
  const int mloc  = lane & 15;
  const int ahalf = (lane >> 4) << 3;   // 0 or 8   (unused)
  const int bbase = (lane >> 4) << 4;   // 0 or 16  (B K-group select)
  const _Float16* arow = ldsA + (size_t)(w * 16 + mloc) * D_DIM;

  float tv[TOPK];   // sorted ascending running top-8
  int   ti[TOPK];
#pragma unroll
  for (int i = 0; i < TOPK; ++i) { tv[i] = -1e30f; ti[i] = 0; }

  const int nPairs = Ncols / 32;
  for (int nt = 0; nt < nPairs; ++nt) {
    const _Float16* brow0 = kn + (size_t)(nt * 32 + mloc) * D_DIM;
    const _Float16* brow1 = brow0 + 16 * D_DIM;
    v8f c0 = {};
    v8f c1 = {};
#pragma unroll 2
    for (int kc = 0; kc < D_DIM; kc += 32) {
      half8 alo = *(const half8*)(arow + kc + bbase);
      half8 ahi = *(const half8*)(arow + kc + bbase + 8);
      half8 b0lo = *(const half8*)(brow0 + kc + bbase);
      half8 b0hi = *(const half8*)(brow0 + kc + bbase + 8);
      half8 b1lo = *(const half8*)(brow1 + kc + bbase);
      half8 b1hi = *(const half8*)(brow1 + kc + bbase + 8);
      v16h a  = __builtin_shufflevector(alo, ahi, 0, 1, 2, 3, 4, 5, 6, 7, 8, 9,
                                        10, 11, 12, 13, 14, 15);
      v16h b0 = __builtin_shufflevector(b0lo, b0hi, 0, 1, 2, 3, 4, 5, 6, 7, 8,
                                        9, 10, 11, 12, 13, 14, 15);
      v16h b1 = __builtin_shufflevector(b1lo, b1hi, 0, 1, 2, 3, 4, 5, 6, 7, 8,
                                        9, 10, 11, 12, 13, 14, 15);
      c0 = __builtin_amdgcn_wmma_f32_16x16x32_f16(false, a, false, b0, (short)0,
                                                  c0, false, false);
      c1 = __builtin_amdgcn_wmma_f32_16x16x32_f16(false, a, false, b1, (short)0,
                                                  c1, false, false);
    }

    // ---- stage both 16x16 f32 tiles as one 16x32 strip in LDS ----
    float* st = stage + w * 512;
    {
      const int col = lane & 15;
      const int rb  = (lane >> 4) * 8;
#pragma unroll
      for (int r = 0; r < 8; ++r) {
        st[(rb + r) * 32 + col] = c0[r];
        st[(rb + r) * 32 + col + 16] = c1[r];
      }
    }
    __syncthreads();
    if (lane < 16) {
      const v4f* rowv = (const v4f*)(st + lane * 32);
#pragma unroll
      for (int j4 = 0; j4 < 8; ++j4) {
        const v4f vv = rowv[j4];
#pragma unroll
        for (int e = 0; e < 4; ++e) {
          const float v = vv[e];
          if (v > tv[0]) {
            const int idx = nt * 32 + j4 * 4 + e;
            int p = 0;
#pragma unroll
            for (int q = 0; q < TOPK - 1; ++q)
              if (tv[q + 1] < v) { tv[q] = tv[q + 1]; ti[q] = ti[q + 1]; p = q + 1; }
            tv[p] = v;
            ti[p] = idx;
          }
        }
      }
    }
    __syncthreads();
  }

  // ---- softmax over top-8 (descending order like jax.lax.top_k) ----
  if (lane < 16) {
    const int row = mBlock + w * 16 + lane;
    const float mx = tv[TOPK - 1];
    float e[TOPK], s = 0.f;
#pragma unroll
    for (int k = 0; k < TOPK; ++k) { e[k] = __expf(tv[TOPK - 1 - k] - mx); s += e[k]; }
    const float inv = 1.f / s;
#pragma unroll
    for (int k = 0; k < TOPK; ++k) {
      topw[(size_t)row * TOPK + k]   = e[k] * inv;
      topi[(size_t)row * TOPK + k]   = ti[TOPK - 1 - k];
      outIdx[(size_t)row * TOPK + k] = (float)ti[TOPK - 1 - k];
    }
  }
}

// ---------------------------------------------------------------------------
// Kernel 3: gather + weighted combine.  One block per query row; each thread
// owns one 16-byte column chunk -> everything is b128 and coalesced.  Source
// tables (16 MB fp32 each) are L2-resident; the ~600 MB of writes are the
// HBM-bound tail of the whole pipeline (~26 us at 23.3 TB/s).
// ---------------------------------------------------------------------------
__global__ __launch_bounds__(256) void gather_combine_kernel(
    const v4f* __restrict__ pc4,      // prompt_components [N, D/4]
    const v4f* __restrict__ keys4,    // component_keys    [N, D/4]
    const float* __restrict__ topw,   // [B, 8]
    const int* __restrict__ topi,     // [B, 8]
    v4f* __restrict__ outP4,          // [B, D/4]
    v4f* __restrict__ outK4) {        // [B, 8, D/4]
  const int b = blockIdx.x;
  const int t = threadIdx.x;          // one float4 per thread (D/4 == 256)
  __shared__ float wsh[TOPK];
  __shared__ int   ish[TOPK];
  if (t < TOPK) {
    wsh[t] = topw[(size_t)b * TOPK + t];
    ish[t] = topi[(size_t)b * TOPK + t];
  }
  __syncthreads();

  v4f acc = {0.f, 0.f, 0.f, 0.f};
#pragma unroll
  for (int k = 0; k < TOPK; ++k) {
    const size_t src = (size_t)ish[k] * (D_DIM / 4) + t;
    acc += wsh[k] * pc4[src];
    outK4[((size_t)b * TOPK + k) * (D_DIM / 4) + t] = keys4[src];
  }
  outP4[(size_t)b * (D_DIM / 4) + t] = acc;
}

// ---------------------------------------------------------------------------
// launch
// ---------------------------------------------------------------------------
extern "C" void kernel_launch(void* const* d_in, const int* in_sizes, int n_in,
                              void* d_out, int out_size, void* d_ws,
                              size_t ws_size, hipStream_t stream) {
  const float* query = (const float*)d_in[0];   // [B, D]
  const float* pc    = (const float*)d_in[1];   // [N, D]
  const float* keys  = (const float*)d_in[2];   // [N, D]
  const float* attn  = (const float*)d_in[3];   // [N, D]

  const int B = in_sizes[0] / D_DIM;            // 16384
  const int N = in_sizes[1] / D_DIM;            // 4096

  // workspace layout
  char* ws = (char*)d_ws;
  float*    attnPartial = (float*)ws;                           // 32*D f32
  float*    attnMean    = attnPartial + 32 * D_DIM;             // D f32
  _Float16* aqn         = (_Float16*)(attnMean + D_DIM);        // B*D f16
  _Float16* knn         = aqn + (size_t)B * D_DIM;              // N*D f16
  float*    topw        = (float*)(knn + (size_t)N * D_DIM);    // B*8 f32
  int*      topi        = (int*)(topw + (size_t)B * TOPK);      // B*8 i32

  // d_out layout: prompts [B*D] | indices-as-float [B*8] | keys [B*8*D]
  float* outP   = (float*)d_out;
  float* outIdx = outP + (size_t)B * D_DIM;
  float* outK   = outIdx + (size_t)B * TOPK;

  // 0) attention mean
  attn_partial_kernel<<<dim3(D_DIM / 256, 32), 256, 0, stream>>>(attn,
                                                                 attnPartial, N);
  attn_reduce_kernel<<<D_DIM / 256, 256, 0, stream>>>(attnPartial, attnMean, N);

  // 1) normalized f16 operands for the WMMA GEMM
  rownorm_f16_kernel<<<B, 256, 0, stream>>>(query, attnMean, aqn);
  rownorm_f16_kernel<<<N, 256, 0, stream>>>(keys, nullptr, knn);

  // 2) fused cosine-sim GEMM + top-8 + softmax (272 KB dynamic LDS / WGP)
  const size_t smemBytes = (size_t)128 * D_DIM * 2 + 8 * 512 * 4;  // 278528
  (void)hipFuncSetAttribute((const void*)sims_topk_kernel,
                            hipFuncAttributeMaxDynamicSharedMemorySize,
                            (int)smemBytes);
  sims_topk_kernel<<<B / 128, 256, smemBytes, stream>>>(aqn, knn, topw, topi,
                                                        outIdx, N);

  // 3) gather + weighted combine (bandwidth-bound output stage)
  gather_combine_kernel<<<B, 256, 0, stream>>>((const v4f*)pc,
                                               (const v4f*)keys, topw, topi,
                                               (v4f*)outP, (v4f*)outK);
}
